// RSSMTransition_84000970375358
// MI455X (gfx1250) — compile-verified, matrix operand order checked by reading, not run on hardware
//
#include <hip/hip_runtime.h>
#include <stdint.h>

// ---------------------------------------------------------------------------
// Problem constants (from reference)
// ---------------------------------------------------------------------------
#define BN      16384          // B*N rows
#define AACT    20             // actions
#define NAGENT  8
#define STOCH   1024
#define DETER   2048
#define HIDDEN  1024
#define NCAT    32
#define NCLS    32
#define GIN     1184           // NAGENT*AACT + STOCH (true K)
#define GINP    1216           // padded to multiple of 64
#define AIN     1044           // AACT + STOCH (true K)
#define AINP    1088           // padded to multiple of 64
#define G3      6144           // 3*DETER
#define CONCAT  3072           // DETER + STOCH (agent latent input)

#define KT      64             // GEMM K tile
#define APITCH  72             // A LDS row pitch (bf16 elems), bank-spread
#define BPITCH  132            // B LDS row pitch (dwords), bank-spread

typedef __bf16 bf16;
typedef float v8f   __attribute__((ext_vector_type(8)));
typedef bf16  v16bf __attribute__((ext_vector_type(16)));
typedef int   v4i   __attribute__((ext_vector_type(4)));
typedef v4i __attribute__((address_space(1)))* gbuf4_t;   // global int4*
typedef v4i __attribute__((address_space(3)))* lbuf4_t;   // LDS    int4*

// async load->LDS builtin is probe-confirmed on this toolchain; wait falls
// back to inline asm if its builtin is absent.
#if defined(__has_builtin)
#if __has_builtin(__builtin_amdgcn_global_load_async_to_lds_b128)
#define USE_ASYNC 1
#if __has_builtin(__builtin_amdgcn_s_wait_asynccnt)
#define WAIT_ASYNC() __builtin_amdgcn_s_wait_asynccnt(0)
#else
#define WAIT_ASYNC() asm volatile("s_wait_asynccnt 0x0" ::: "memory")
#endif
#endif
#endif
#ifndef USE_ASYNC
#define USE_ASYNC 0
#endif

__device__ __forceinline__ bf16 f2bf(float f) {
  uint32_t u = __builtin_bit_cast(uint32_t, f);
  u += 0x7FFFu + ((u >> 16) & 1u);            // round-to-nearest-even
  uint16_t h = (uint16_t)(u >> 16);
  return __builtin_bit_cast(bf16, h);
}
__device__ __forceinline__ float bf2f(bf16 x) {
  uint32_t u = (uint32_t)__builtin_bit_cast(uint16_t, x) << 16;
  return __builtin_bit_cast(float, u);
}

// ---------------------------------------------------------------------------
// bf16 WMMA GEMM:  C = act(A[M,K] * B[K,N] + bias),  M=BN fixed by grid.y
// Block 128x128, Ktile 64, double-buffered LDS, one barrier per K-tile.
// 8 waves (2M x 4N), wave tile 64x32 -> 4x2 wmma 16x16x32 tiles, f32 acc.
// A staged row-major (pitch APITCH), async-copied to LDS (ASYNCcnt);
// B staged pair-packed: Bp[kp][n] dword = {B[2kp+1][n], B[2kp][n]} matching
// the 16-bit WMMA operand K-pair layout.
// Requires N%128==0, K%64==0 (caller pads K).
// ---------------------------------------------------------------------------
union FragA { uint32_t u[8]; v16bf v; };
union FragC { float    f[8]; v8f   v; };

__global__ __launch_bounds__(256)
void gemm_bf16_wmma(const bf16* __restrict__ A, const bf16* __restrict__ B,
                    const float* __restrict__ bias,
                    float* __restrict__ Cf, bf16* __restrict__ Cb,
                    int N, int K, int lda, int ldb,
                    int ldcf, int ldcb, int relu)
{
  __shared__ bf16     As[2][128 * APITCH];   // 2 x 18 KB
  __shared__ uint32_t Bp[2][32 * BPITCH];    // 2 x 16.5 KB

  const int tid  = threadIdx.x;
  const int lane = tid & 31;
  const int wave = tid >> 5;
  const int wm   = wave >> 2;        // 0..1
  const int wn   = wave & 3;         // 0..3
  const int m0   = blockIdx.y * 128;
  const int n0   = blockIdx.x * 128;

  const int lr  = lane & 15;         // row (A) / col (B,C) within 16
  const int kb  = (lane >> 4) * 8;   // K elem sub-base (A frag) == C row sub-base
  const int kb2 = (lane >> 4) * 4;   // K pair sub-base (B frag)

  // staging coordinates
  const int ar  = tid >> 1;                  // A: tile row 0..127
  const int ah  = (tid & 1) * 32;            // A: elem offset (32 elems = 64B)
  const int bkp = tid >> 3;                  // B: pair-row 0..31
  const int bcc = (tid & 7) * 16;            // B: col offset 0..112

  const int nk = K / KT;

  FragC acc[4][2];
#pragma unroll
  for (int mi = 0; mi < 4; ++mi)
#pragma unroll
    for (int ni = 0; ni < 2; ++ni)
#pragma unroll
      for (int r = 0; r < 8; ++r) acc[mi][ni].f[r] = 0.0f;

  uint4    areg[4];
  uint32_t blo[8], bhi[8];

  auto loadB = [&](int t) {   // global -> regs (two K-adjacent rows, 16 cols)
    const bf16* s0 = B + (size_t)(t * KT + 2 * bkp) * ldb + n0 + bcc;
    const uint4* p0 = (const uint4*)s0;
    const uint4* p1 = (const uint4*)(s0 + ldb);
    uint4 q0 = p0[0], q1 = p0[1], q2 = p1[0], q3 = p1[1];
    blo[0] = q0.x; blo[1] = q0.y; blo[2] = q0.z; blo[3] = q0.w;
    blo[4] = q1.x; blo[5] = q1.y; blo[6] = q1.z; blo[7] = q1.w;
    bhi[0] = q2.x; bhi[1] = q2.y; bhi[2] = q2.z; bhi[3] = q2.w;
    bhi[4] = q3.x; bhi[5] = q3.y; bhi[6] = q3.z; bhi[7] = q3.w;
  };
  auto storeB = [&](int buf) { // regs -> LDS pair-packed
    uint32_t o[16];
#pragma unroll
    for (int c = 0; c < 8; ++c) {
      const uint32_t lo = blo[c], hi = bhi[c];
      o[2 * c]     = (lo & 0xFFFFu) | (hi << 16);
      o[2 * c + 1] = (lo >> 16)     | (hi & 0xFFFF0000u);
    }
    uint4* dst = (uint4*)&Bp[buf][bkp * BPITCH + bcc];
#pragma unroll
    for (int q = 0; q < 4; ++q) {
      uint4 v; v.x = o[4*q]; v.y = o[4*q+1]; v.z = o[4*q+2]; v.w = o[4*q+3];
      dst[q] = v;
    }
  };
#if USE_ASYNC
  auto issueA = [&](int t, int buf) {   // async global -> LDS (ASYNCcnt)
    const bf16* g = A + (size_t)(m0 + ar) * lda + t * KT + ah;
    bf16*       l = &As[buf][ar * APITCH + ah];
#pragma unroll
    for (int q = 0; q < 4; ++q)
      __builtin_amdgcn_global_load_async_to_lds_b128(
          (gbuf4_t)(g + q * 8), (lbuf4_t)(l + q * 8), 0, 0);
  };
#else
  auto loadA = [&](int t) {
    const uint4* src = (const uint4*)(A + (size_t)(m0 + ar) * lda + t * KT + ah);
    areg[0] = src[0]; areg[1] = src[1]; areg[2] = src[2]; areg[3] = src[3];
  };
  auto storeA = [&](int buf) {
    uint4* dst = (uint4*)&As[buf][ar * APITCH + ah];
    dst[0] = areg[0]; dst[1] = areg[1]; dst[2] = areg[2]; dst[3] = areg[3];
  };
#endif

  auto compute = [&](int buf) {
#pragma unroll
    for (int s = 0; s < 2; ++s) {       // two 32-deep sub-steps per tile
      FragA a[4], b[2];
#pragma unroll
      for (int mi = 0; mi < 4; ++mi) {
        const bf16* p = &As[buf][(wm * 64 + mi * 16 + lr) * APITCH + s * 32 + kb];
#pragma unroll
        for (int j = 0; j < 4; ++j) a[mi].u[j]     = *(const uint32_t*)(p + 2 * j);
#pragma unroll
        for (int j = 0; j < 4; ++j) a[mi].u[4 + j] = *(const uint32_t*)(p + 16 + 2 * j);
      }
#pragma unroll
      for (int ni = 0; ni < 2; ++ni) {
        const int col = wn * 32 + ni * 16 + lr;
        const int kp0 = s * 16 + kb2;
#pragma unroll
        for (int j = 0; j < 4; ++j) b[ni].u[j]     = Bp[buf][(kp0 + j) * BPITCH + col];
#pragma unroll
        for (int j = 0; j < 4; ++j) b[ni].u[4 + j] = Bp[buf][(kp0 + 8 + j) * BPITCH + col];
      }
#pragma unroll
      for (int mi = 0; mi < 4; ++mi)
#pragma unroll
        for (int ni = 0; ni < 2; ++ni)
          acc[mi][ni].v = __builtin_amdgcn_wmma_f32_16x16x32_bf16(
              false, a[mi].v, false, b[ni].v, (short)0, acc[mi][ni].v, false, false);
    }
  };

  // ---- pipelined main loop: stage t+1 while computing t, 1 barrier/iter ----
  loadB(0);
#if USE_ASYNC
  issueA(0, 0);
#else
  loadA(0);
  storeA(0);
#endif
  storeB(0);

  for (int t = 0; t < nk; ++t) {
    const int cur = t & 1, nxt = (t + 1) & 1;
    const bool more = (t + 1 < nk);
    if (more) {
      loadB(t + 1);
#if !USE_ASYNC
      loadA(t + 1);
#endif
      if (t + 2 < nk)   // global_prefetch_b8 hint for tile t+2
        __builtin_prefetch((const void*)(B + (size_t)((t + 2) * KT + 2 * bkp) * ldb + n0 + bcc), 0, 1);
    }
#if USE_ASYNC
    WAIT_ASYNC();                          // A tile t landed in LDS
#endif
    __syncthreads();                       // tile t visible; prev readers done
    if (more) {
#if USE_ASYNC
      issueA(t + 1, nxt);
#else
      storeA(nxt);
#endif
      storeB(nxt);
    }
    compute(cur);
  }

  // ---- epilogue: bias (+ReLU), write f32 and/or bf16 ----
#pragma unroll
  for (int mi = 0; mi < 4; ++mi) {
#pragma unroll
    for (int ni = 0; ni < 2; ++ni) {
      const int col = n0 + wn * 32 + ni * 16 + lr;
      const float bv = bias ? bias[col] : 0.0f;
#pragma unroll
      for (int r = 0; r < 8; ++r) {
        const int row = m0 + wm * 64 + mi * 16 + kb + r;
        float v = acc[mi][ni].f[r] + bv;
        if (relu) v = v > 0.0f ? v : 0.0f;
        if (Cf) Cf[(size_t)row * ldcf + col] = v;
        if (Cb) Cb[(size_t)row * ldcb + col] = f2bf(v);
      }
    }
  }
}

// ---------------------------------------------------------------------------
// Weight convert: fp32 -> bf16, optional transpose, optional zero row padding
// ---------------------------------------------------------------------------
__global__ void wconv_kernel(const float* __restrict__ src, bf16* __restrict__ dst,
                             int rows, int cols, int dstRows, int transp)
{
  const int total = transp ? rows * cols : dstRows * cols;
  for (int idx = blockIdx.x * blockDim.x + threadIdx.x; idx < total;
       idx += gridDim.x * blockDim.x) {
    const int r = idx / cols, c = idx % cols;
    if (transp) dst[(size_t)c * rows + r] = f2bf(src[idx]);
    else        dst[idx] = (r < rows) ? f2bf(src[(size_t)r * cols + c]) : f2bf(0.0f);
  }
}

__global__ void cvt_f32_bf16_kernel(const float* __restrict__ src,
                                    bf16* __restrict__ dst, int n)
{
  for (int i = blockIdx.x * blockDim.x + threadIdx.x; i < n;
       i += gridDim.x * blockDim.x)
    dst[i] = f2bf(src[i]);
}

// ---------------------------------------------------------------------------
// Build bf16 input matrices (K-padded with zeros):
//  Xg[BN,GINP] = [onehot(pa,20) | onehot(poa_j,20)x7 | global_stoch | 0-pad]
//  Xa[BN,AINP] = [onehot(pa,20) | agent_stoch | 0-pad]
// ---------------------------------------------------------------------------
__global__ void build_inputs_kernel(const int* __restrict__ pa,
                                    const int* __restrict__ poa,
                                    const float* __restrict__ gst,
                                    const float* __restrict__ ast,
                                    bf16* __restrict__ Xg, bf16* __restrict__ Xa)
{
  const int row = blockIdx.x;
  const int a0  = pa[row];
  bf16* xg = Xg + (size_t)row * GINP;
  bf16* xa = Xa + (size_t)row * AINP;
  for (int c = threadIdx.x; c < GINP; c += blockDim.x) {
    float v;
    if (c < AACT) v = (c == a0) ? 1.0f : 0.0f;
    else if (c < AACT * NAGENT) {
      const int j = (c - AACT) / AACT, cl = (c - AACT) % AACT;
      v = (poa[(size_t)row * (NAGENT - 1) + j] == cl) ? 1.0f : 0.0f;
    } else if (c < GIN) v = gst[(size_t)row * STOCH + (c - AACT * NAGENT)];
    else v = 0.0f;
    xg[c] = f2bf(v);
  }
  for (int c = threadIdx.x; c < AINP; c += blockDim.x) {
    float v;
    if (c < AACT)      v = (c == a0) ? 1.0f : 0.0f;
    else if (c < AIN)  v = ast[(size_t)row * STOCH + (c - AACT)];
    else               v = 0.0f;
    xa[c] = f2bf(v);
  }
}

// ---------------------------------------------------------------------------
// GRU gate combine:  h' = (1-z)*n + z*h, gates read as bf16 pre-activations
// (f32-accumulated in the GEMM, bf16-stored to halve gate HBM traffic).
// ---------------------------------------------------------------------------
__global__ void gru_combine_kernel(const bf16* __restrict__ gi,
                                   const bf16* __restrict__ gh,
                                   const float* __restrict__ h,
                                   float* __restrict__ outf,
                                   bf16* __restrict__ outb, int bpitch)
{
  const int idx = blockIdx.x * blockDim.x + threadIdx.x;
  if (idx >= BN * DETER) return;
  const int m = idx / DETER, c = idx % DETER;
  const size_t base = (size_t)m * G3;
  const float ir = bf2f(gi[base + c]), iz = bf2f(gi[base + DETER + c]),
              in = bf2f(gi[base + 2 * DETER + c]);
  const float hr = bf2f(gh[base + c]), hz = bf2f(gh[base + DETER + c]),
              hn = bf2f(gh[base + 2 * DETER + c]);
  const float r = 1.0f / (1.0f + __expf(-(ir + hr)));
  const float z = 1.0f / (1.0f + __expf(-(iz + hz)));
  const float n = tanhf(in + r * hn);
  const float v = (1.0f - z) * n + z * h[idx];
  outf[idx] = v;
  if (outb) outb[(size_t)m * bpitch + c] = f2bf(v);
}

// ---------------------------------------------------------------------------
// Discrete latent: softmax over 32 classes, Gumbel-max categorical sample
// (deterministic splitmix64 counter RNG), straight-through value.
// ---------------------------------------------------------------------------
__device__ __forceinline__ float u01(uint64_t x) {
  x += 0x9E3779B97F4A7C15ull;
  x = (x ^ (x >> 30)) * 0xBF58476D1CE4E5B9ull;
  x = (x ^ (x >> 27)) * 0x94D049BB133111EBull;
  x ^= x >> 31;
  return ((float)(uint32_t)(x >> 40) + 0.5f) * (1.0f / 16777216.0f);
}

__global__ void latent_kernel(const float* __restrict__ logits,
                              float* __restrict__ stoch,
                              bf16* __restrict__ bfout, int bpitch, int bcoloff,
                              unsigned seed)
{
  const int idx = blockIdx.x * blockDim.x + threadIdx.x;
  if (idx >= BN * NCAT) return;
  const int row = idx >> 5, cat = idx & 31;
  const float* lg = logits + (size_t)row * (NCAT * NCLS) + cat * NCLS;
  float l[NCLS], p[NCLS];
  float mx = -1e30f;
#pragma unroll
  for (int c = 0; c < NCLS; ++c) { l[c] = lg[c]; mx = l[c] > mx ? l[c] : mx; }
  float s = 0.0f;
#pragma unroll
  for (int c = 0; c < NCLS; ++c) { p[c] = __expf(l[c] - mx); s += p[c]; }
  const float inv = 1.0f / s;
#pragma unroll
  for (int c = 0; c < NCLS; ++c) p[c] *= inv;
  int arg = 0; float best = -1e30f;
#pragma unroll
  for (int c = 0; c < NCLS; ++c) {
    const uint64_t x = ((uint64_t)seed << 40) ^ ((uint64_t)idx << 6) ^ (uint64_t)c;
    const float g = -__logf(-__logf(u01(x)));
    const float sc = l[c] + g;
    if (sc > best) { best = sc; arg = c; }
  }
  float* so = stoch + (size_t)row * (NCAT * NCLS) + cat * NCLS;
#pragma unroll
  for (int c = 0; c < NCLS; ++c) {
    const float oh = (c == arg) ? 1.0f : 0.0f;
    const float v = (oh + p[c]) - p[c];
    so[c] = v;
    if (bfout) bfout[(size_t)row * bpitch + bcoloff + cat * NCLS + c] = f2bf(v);
  }
}

// ---------------------------------------------------------------------------
// Host orchestration
// ---------------------------------------------------------------------------
static inline int nblk(long long total) { return (int)((total + 255) / 256); }

extern "C" void kernel_launch(void* const* d_in, const int* in_sizes, int n_in,
                              void* d_out, int out_size, void* d_ws, size_t ws_size,
                              hipStream_t stream)
{
  (void)in_sizes; (void)n_in; (void)out_size; (void)ws_size;
  const int*   prev_a  = (const int*)  d_in[0];
  const int*   prev_oa = (const int*)  d_in[1];
  const float* gstoch  = (const float*)d_in[2];
  const float* gdeter  = (const float*)d_in[3];
  const float* astoch  = (const float*)d_in[4];
  const float* adeter  = (const float*)d_in[5];
  const float* gW1 = (const float*)d_in[6];  const float* gb1 = (const float*)d_in[7];
  const float* gW2 = (const float*)d_in[8];  const float* gb2 = (const float*)d_in[9];
  const float* gWi = (const float*)d_in[10]; const float* gWh = (const float*)d_in[11];
  const float* gbi = (const float*)d_in[12]; const float* gbh = (const float*)d_in[13];
  const float* gpW1 = (const float*)d_in[14]; const float* gpb1 = (const float*)d_in[15];
  const float* gpW2 = (const float*)d_in[16]; const float* gpb2 = (const float*)d_in[17];
  const float* aW1 = (const float*)d_in[18]; const float* ab1 = (const float*)d_in[19];
  const float* aW2 = (const float*)d_in[20]; const float* ab2 = (const float*)d_in[21];
  const float* aWi = (const float*)d_in[22]; const float* aWh = (const float*)d_in[23];
  const float* abi = (const float*)d_in[24]; const float* abh = (const float*)d_in[25];
  const float* apW1 = (const float*)d_in[26]; const float* apb1 = (const float*)d_in[27];
  const float* apW2 = (const float*)d_in[28]; const float* apb2 = (const float*)d_in[29];

  float* out = (float*)d_out;
  float* o_glogits = out;
  float* o_gstoch  = out + (size_t)BN * 1024;
  float* o_gdet    = out + (size_t)BN * 2048;
  float* o_alogits = out + (size_t)BN * 4096;
  float* o_astoch  = out + (size_t)BN * 5120;
  float* o_adet    = out + (size_t)BN * 6144;

  // ---- workspace carve-up (256B aligned) ----
  char* ws = (char*)d_ws;
  size_t off = 0;
  auto ab = [&](size_t n) -> bf16* {
    off = (off + 255) & ~(size_t)255; bf16* p = (bf16*)(ws + off); off += n * 2; return p;
  };
  bf16* WgW1  = ab((size_t)GINP * HIDDEN);   // zero-padded rows GIN..GINP
  bf16* WgW2  = ab((size_t)HIDDEN * HIDDEN);
  bf16* WgWiT = ab((size_t)HIDDEN * G3);     // gWi^T [1024,6144]
  bf16* WgWhT = ab((size_t)DETER  * G3);     // gWh^T [2048,6144]
  bf16* WgpW1 = ab((size_t)DETER  * HIDDEN);
  bf16* WgpW2 = ab((size_t)HIDDEN * (NCAT * NCLS));
  bf16* WaW1  = ab((size_t)AINP * HIDDEN);   // zero-padded rows AIN..AINP
  bf16* WaW2  = ab((size_t)HIDDEN * HIDDEN);
  bf16* WaWiT = ab((size_t)HIDDEN * G3);
  bf16* WaWhT = ab((size_t)DETER  * G3);
  bf16* WapW1 = ab((size_t)CONCAT * HIDDEN);
  bf16* WapW2 = ab((size_t)HIDDEN * (NCAT * NCLS));
  bf16* Xg   = ab((size_t)BN * GINP);
  bf16* Xa   = ab((size_t)BN * AINP);
  bf16* BufA = ab((size_t)BN * HIDDEN);
  bf16* BufB = ab((size_t)BN * HIDDEN);
  bf16* Hbf  = ab((size_t)BN * DETER);       // bf16 deter state / deter output
  bf16* Xap  = ab((size_t)BN * CONCAT);      // [a_det | g_stoch] bf16
  bf16* GI   = ab((size_t)BN * G3);          // gi gates bf16 (reused per branch)
  bf16* GH   = ab((size_t)BN * G3);          // gh gates bf16 (reused per branch)

  auto gemm = [&](const bf16* A, const bf16* B, const float* bias,
                  float* Cf, bf16* Cb, int N, int K, int lda, int ldb,
                  int ldcf, int ldcb, int relu) {
    dim3 g((unsigned)(N / 128), (unsigned)(BN / 128));
    gemm_bf16_wmma<<<g, 256, 0, stream>>>(A, B, bias, Cf, Cb,
                                          N, K, lda, ldb, ldcf, ldcb, relu);
  };
  auto wconv = [&](const float* s, bf16* d, int rows, int cols, int dstRows, int tr) {
    const long long tot = (long long)(tr ? rows : dstRows) * cols;
    wconv_kernel<<<nblk(tot), 256, 0, stream>>>(s, d, rows, cols, dstRows, tr);
  };

  // ---- 0) weight conversion (fp32 -> bf16, transpose GRU weights) ----
  wconv(gW1,  WgW1,  GIN,    HIDDEN, GINP,   0);
  wconv(gW2,  WgW2,  HIDDEN, HIDDEN, HIDDEN, 0);
  wconv(gWi,  WgWiT, G3,     HIDDEN, 0,      1);   // -> [1024,6144]
  wconv(gWh,  WgWhT, G3,     DETER,  0,      1);   // -> [2048,6144]
  wconv(gpW1, WgpW1, DETER,  HIDDEN, DETER,  0);
  wconv(gpW2, WgpW2, HIDDEN, NCAT * NCLS, HIDDEN, 0);
  wconv(aW1,  WaW1,  AIN,    HIDDEN, AINP,   0);   // pad K 1044->1088
  wconv(aW2,  WaW2,  HIDDEN, HIDDEN, HIDDEN, 0);
  wconv(aWi,  WaWiT, G3,     HIDDEN, 0,      1);
  wconv(aWh,  WaWhT, G3,     DETER,  0,      1);
  wconv(apW1, WapW1, CONCAT, HIDDEN, CONCAT, 0);
  wconv(apW2, WapW2, HIDDEN, NCAT * NCLS, HIDDEN, 0);

  // ---- 1) one-hot + concat bf16 inputs ----
  build_inputs_kernel<<<BN, 256, 0, stream>>>(prev_a, prev_oa, gstoch, astoch, Xg, Xa);

  // ---- 2) global branch ----
  cvt_f32_bf16_kernel<<<nblk((long long)BN * DETER), 256, 0, stream>>>(gdeter, Hbf, BN * DETER);
  gemm(Xg,   WgW1,  gb1, nullptr, BufA, HIDDEN, GINP,   GINP,   HIDDEN, 0, HIDDEN, 1); // h1
  gemm(BufA, WgW2,  gb2, nullptr, BufB, HIDDEN, HIDDEN, HIDDEN, HIDDEN, 0, HIDDEN, 1); // x
  gemm(BufB, WgWiT, gbi, nullptr, GI, G3, HIDDEN, HIDDEN, G3, 0, G3, 0);               // gi (bf16)
  gemm(Hbf,  WgWhT, gbh, nullptr, GH, G3, DETER,  DETER,  G3, 0, G3, 0);               // gh (bf16)
  gru_combine_kernel<<<nblk((long long)BN * DETER), 256, 0, stream>>>(
      GI, GH, gdeter, o_gdet, Hbf, DETER);                                             // g_det (Hbf reused)
  gemm(Hbf,  WgpW1, gpb1, nullptr, BufB, HIDDEN, DETER,  DETER,  HIDDEN, 0, HIDDEN, 1);
  gemm(BufB, WgpW2, gpb2, o_glogits, nullptr, NCAT * NCLS, HIDDEN, HIDDEN, NCAT * NCLS,
       NCAT * NCLS, 0, 0);                                                             // g_logits
  latent_kernel<<<nblk((long long)BN * NCAT), 256, 0, stream>>>(
      o_glogits, o_gstoch, Xap, CONCAT, DETER, 42u);  // g_stoch -> out + Xap[:,2048:]

  // ---- 3) agent branch ----
  gemm(Xa,   WaW1,  ab1, nullptr, BufA, HIDDEN, AINP,   AINP,   HIDDEN, 0, HIDDEN, 1);
  gemm(BufA, WaW2,  ab2, nullptr, BufB, HIDDEN, HIDDEN, HIDDEN, HIDDEN, 0, HIDDEN, 1);
  gemm(BufB, WaWiT, abi, nullptr, GI, G3, HIDDEN, HIDDEN, G3, 0, G3, 0);
  cvt_f32_bf16_kernel<<<nblk((long long)BN * DETER), 256, 0, stream>>>(adeter, Hbf, BN * DETER);
  gemm(Hbf,  WaWhT, abh, nullptr, GH, G3, DETER, DETER, G3, 0, G3, 0);
  gru_combine_kernel<<<nblk((long long)BN * DETER), 256, 0, stream>>>(
      GI, GH, adeter, o_adet, Xap, CONCAT);            // a_det -> out + Xap[:,0:2048]
  gemm(Xap,  WapW1, apb1, nullptr, BufA, HIDDEN, CONCAT, CONCAT, HIDDEN, 0, HIDDEN, 1);
  gemm(BufA, WapW2, apb2, o_alogits, nullptr, NCAT * NCLS, HIDDEN, HIDDEN, NCAT * NCLS,
       NCAT * NCLS, 0, 0);
  latent_kernel<<<nblk((long long)BN * NCAT), 256, 0, stream>>>(
      o_alogits, o_astoch, nullptr, 0, 0, 43u);
}
